// CoreTokenAttention_test_61014305407020
// MI455X (gfx1250) — compile-verified, matrix operand order checked by reading, not run on hardware
//
#include <hip/hip_runtime.h>
#include <hip/hip_bf16.h>

// ---------------- problem constants ----------------
#define S_LEN   4096
#define D_MODEL 1024
#define NHEAD   16
#define HD      64
#define R_IN    3686
#define R_PAD   3712          // multiple of 32
#define R_OUT_N 3317
#define NEG_INF_SCORE (-1e10f)
#define SCALE_INV 0.125f      // 1/sqrt(64)

typedef __attribute__((ext_vector_type(16))) __bf16 bf16x16;
typedef __attribute__((ext_vector_type(8)))  __bf16 bf16x8;
typedef __attribute__((ext_vector_type(4)))  __bf16 bf16x4;
typedef __attribute__((ext_vector_type(8)))  float  f32x8;
typedef __attribute__((ext_vector_type(4)))  unsigned int u32x4;
typedef __attribute__((ext_vector_type(8)))  int i32x8;
typedef __attribute__((ext_vector_type(4)))  int i32x4;

// ---------------- helpers ----------------
__device__ inline f32x8 zero8() {
    f32x8 z;
#pragma unroll
    for (int i = 0; i < 8; ++i) z[i] = 0.0f;
    return z;
}

// 16x32 bf16 A/B fragment (wave32 layout, CDNA5 ISA §7.12.2) from a
// row-major bf16 row: lane's elements are base[half*8..+7], base[16+half*8..+7].
__device__ inline bf16x16 frag_from_bf16(const __bf16* base, int half) {
    bf16x8 lo = *(const bf16x8*)(base + half * 8);
    bf16x8 hi = *(const bf16x8*)(base + 16 + half * 8);
    bf16x16 r;
#pragma unroll
    for (int i = 0; i < 8; ++i) { r[i] = lo[i]; r[8 + i] = hi[i]; }
    return r;
}

__device__ inline f32x8 wmma_bf16(bf16x16 a, bf16x16 b, f32x8 c) {
    return __builtin_amdgcn_wmma_f32_16x16x32_bf16(false, a, false, b, (short)0, c, false, false);
}

// Tensor Data Mover: 2D (or 1D if tile_d1==0) bf16 tile Global -> LDS.
// D# per CDNA5 ISA; tensor dims == tile dims (no OOB),
// data_size = 2 bytes, stride0 in elements. Issue from ONE wave; completion
// via TENSORcnt of the issuing wave, published to the block by a barrier.
__device__ inline void tdm_load_2d_bf16(void* lds_ptr, const void* gptr,
                                        unsigned tile_d0, unsigned tile_d1,
                                        unsigned long long stride0) {
    unsigned long long ga = (unsigned long long)(uintptr_t)gptr;
    unsigned td0 = tile_d0;
    unsigned td1 = tile_d1 ? tile_d1 : 1u;
    u32x4 g0;
    g0[0] = 1u;                                    // count=1, user mode
    g0[1] = (unsigned)(uintptr_t)lds_ptr;          // lds_addr
    g0[2] = (unsigned)(ga & 0xffffffffull);        // global_addr[31:0]
    g0[3] = (unsigned)((ga >> 32) & 0x01ffffffull) | (2u << 30); // addr[56:32] | type=2
    i32x8 g1;
    g1[0] = (int)(1u << 16);                       // workgroup_mask=0, data_size=1 (2B)
    g1[1] = (int)((td0 & 0xffffu) << 16);          // tensor_dim0[15:0]
    g1[2] = (int)((td0 >> 16) | ((td1 & 0xffffu) << 16));     // dim0[31:16] | dim1[15:0]
    g1[3] = (int)((td1 >> 16) | ((tile_d0 & 0xffffu) << 16)); // dim1[31:16] | tile_dim0
    g1[4] = (int)(tile_d1 & 0xffffu);              // tile_dim1 (0 = 1D), tile_dim2=0
    g1[5] = (int)(stride0 & 0xffffffffull);        // tensor_dim0_stride[31:0]
    g1[6] = (int)((stride0 >> 32) & 0xffffull);
    g1[7] = 0;
    i32x4 z4 = {0, 0, 0, 0};
#if defined(__clang_major__) && __clang_major__ >= 23
    i32x8 z8 = {0, 0, 0, 0, 0, 0, 0, 0};
    __builtin_amdgcn_tensor_load_to_lds(g0, g1, z4, z4, z8, 0);
#else
    __builtin_amdgcn_tensor_load_to_lds(g0, g1, z4, z4, 0);
#endif
}

// ---------------- kernels ----------------

__global__ __launch_bounds__(256) void k_zero_f32(float* p, int n) {
    int i = blockIdx.x * blockDim.x + threadIdx.x;
    if (i < n) p[i] = 0.0f;
}

// one-time f32 -> bf16 array conversion (n divisible by 4)
__global__ __launch_bounds__(256) void k_cvt_bf16(const float* __restrict__ in,
                                                  __bf16* __restrict__ out, int n4) {
    int i = blockIdx.x * blockDim.x + threadIdx.x;
    if (i < n4) {
        float4 v = ((const float4*)in)[i];
        bf16x4 o;
        o[0] = (__bf16)v.x; o[1] = (__bf16)v.y; o[2] = (__bf16)v.z; o[3] = (__bf16)v.w;
        ((bf16x4*)out)[i] = o;
    }
}

// GEMM core: P = A(bf16)[4096,1024] @ W(bf16)^T + bias.
// Block = 8 waves x 32 rows = 256 rows, 64 cols. Weight tile (64 rows x 32 k)
// double-buffered in LDS via the TDM: tile i+1 is in flight while all 8 waves
// consume tile i; one barrier + one s_wait_tensorcnt per k-step.
#define GEMM_BODY(STORE_STMT)                                                           \
    __shared__ __align__(16) __bf16 Btile[2][64 * 32];                                  \
    int wv = threadIdx.x >> 5, lane = threadIdx.x & 31;                                 \
    int n15 = lane & 15, half = lane >> 4;                                              \
    int m0 = blockIdx.y * 256 + wv * 32;                                                \
    int n0 = blockIdx.x * 64;                                                           \
    const __bf16* arow0 = A + (size_t)(m0 + n15) * D_MODEL;                             \
    const __bf16* arow1 = A + (size_t)(m0 + 16 + n15) * D_MODEL;                        \
    const __bf16* wbase = W + (size_t)n0 * D_MODEL;                                     \
    f32x8 acc[2][4];                                                                    \
    _Pragma("unroll") for (int u = 0; u < 2; ++u)                                       \
        _Pragma("unroll") for (int t = 0; t < 4; ++t) acc[u][t] = zero8();              \
    if (wv == 0) tdm_load_2d_bf16(&Btile[0][0], wbase, 32, 64, D_MODEL);                \
    __builtin_amdgcn_s_wait_tensorcnt(0);                                               \
    __syncthreads();                                                                    \
    for (int it = 0; it < D_MODEL / 32; ++it) {                                         \
        int k0 = it * 32;                                                               \
        if (it + 1 < D_MODEL / 32 && wv == 0)                                           \
            tdm_load_2d_bf16(&Btile[(it + 1) & 1][0], wbase + k0 + 32, 32, 64, D_MODEL);\
        if (k0 + 32 < D_MODEL) __builtin_prefetch(arow0 + k0 + 32, 0, 3);               \
        bf16x16 a0 = frag_from_bf16(arow0 + k0, half);                                  \
        bf16x16 a1 = frag_from_bf16(arow1 + k0, half);                                  \
        const __bf16* bt = &Btile[it & 1][0];                                           \
        _Pragma("unroll") for (int t = 0; t < 4; ++t) {                                 \
            bf16x16 bf = frag_from_bf16(bt + (t * 16 + n15) * 32, half);                \
            acc[0][t] = wmma_bf16(a0, bf, acc[0][t]);                                   \
            acc[1][t] = wmma_bf16(a1, bf, acc[1][t]);                                   \
        }                                                                               \
        __builtin_amdgcn_s_wait_tensorcnt(0);                                           \
        __syncthreads();                                                                \
    }                                                                                   \
    _Pragma("unroll") for (int u = 0; u < 2; ++u)                                       \
        _Pragma("unroll") for (int t = 0; t < 4; ++t) {                                 \
            int col = n0 + t * 16 + n15;                                                \
            float bv = bias[col];                                                       \
            _Pragma("unroll") for (int v = 0; v < 8; ++v) {                             \
                int row = m0 + u * 16 + v + 8 * half;                                   \
                STORE_STMT;                                                             \
            }                                                                           \
        }

__global__ __launch_bounds__(256) void k_gemm_qkv(const __bf16* __restrict__ A,
                                                  const __bf16* __restrict__ W,
                                                  const float* __restrict__ bias,
                                                  __bf16* __restrict__ out) {
    GEMM_BODY(out[(size_t)row * D_MODEL + col] = (__bf16)(acc[u][t][v] + bv))
}

__global__ __launch_bounds__(256) void k_gemm_out(const __bf16* __restrict__ A,
                                                  const __bf16* __restrict__ W,
                                                  const float* __restrict__ bias,
                                                  float* __restrict__ out) {
    GEMM_BODY(out[(size_t)row * D_MODEL + col] = acc[u][t][v] + bv)
}

// Gather K_p (row-major per head), V_p^T (d-major), gathered mask; pad r>=R_IN.
__global__ __launch_bounds__(256) void k_gather(const __bf16* __restrict__ Kb,
                                                const __bf16* __restrict__ Vb,
                                                const int* __restrict__ topk,
                                                const int* __restrict__ mask,
                                                __bf16* __restrict__ Kp,
                                                __bf16* __restrict__ VpT,
                                                float* __restrict__ maskok) {
    int t = blockIdx.x * blockDim.x + threadIdx.x;
    if (t >= NHEAD * R_PAD) return;
    int h = t / R_PAD, r = t % R_PAD;
    __bf16* kd = Kp + ((size_t)h * R_PAD + r) * HD;
    if (r < R_IN) {
        int idx = topk[h * R_IN + r];
        const __bf16* ks = Kb + (size_t)idx * D_MODEL + h * HD;
        const __bf16* vs = Vb + (size_t)idx * D_MODEL + h * HD;
#pragma unroll 8
        for (int d = 0; d < HD; ++d) {
            kd[d] = ks[d];
            VpT[((size_t)h * HD + d) * R_PAD + r] = vs[d];
        }
        maskok[h * R_PAD + r] = (float)mask[idx];
    } else {
        __bf16 z = (__bf16)0.0f;
#pragma unroll 8
        for (int d = 0; d < HD; ++d) {
            kd[d] = z;
            VpT[((size_t)h * HD + d) * R_PAD + r] = z;
        }
        maskok[h * R_PAD + r] = 0.0f;
    }
}

// Attention: block = 8 waves on the SAME head (8 adjacent 16-query blocks).
// K/V^T r-tiles double-buffered in LDS via TDM, shared 8-ways.
__global__ __launch_bounds__(256) void k_attn(const __bf16* __restrict__ Qb,
                                              const __bf16* __restrict__ Kp,
                                              const __bf16* __restrict__ VpT,
                                              const float* __restrict__ maskok,
                                              float* __restrict__ imp,
                                              __bf16* __restrict__ ctx) {
    __shared__ __align__(16) __bf16 Ktile[2][32 * 64];   // 2 x 4KB
    __shared__ __align__(16) __bf16 Vtile[2][64 * 32];   // 2 x 4KB
    __shared__ __align__(16) __bf16 pstage[8][16 * 32];  // 1KB per wave
    int wv = threadIdx.x >> 5, lane = threadIdx.x & 31;
    int n15 = lane & 15, half = lane >> 4;
    int h = blockIdx.x >> 5;                 // 16 heads
    int q0 = (((blockIdx.x & 31) << 3) + wv) * 16;
    const int NT = R_PAD / 32;

    const __bf16* qrow = Qb + (size_t)(q0 + n15) * D_MODEL + h * HD;
    bf16x16 aq0 = frag_from_bf16(qrow, half);
    bf16x16 aq1 = frag_from_bf16(qrow + 32, half);

    const __bf16* kbase = Kp + (size_t)h * R_PAD * HD;
    const __bf16* vbase = VpT + (size_t)h * HD * R_PAD;
    const float* mkbase = maskok + (size_t)h * R_PAD;

    float m[8], s[8];
#pragma unroll
    for (int v = 0; v < 8; ++v) { m[v] = -INFINITY; s[v] = 0.0f; }

    // -------- pass 1: online row max + sum (K tiles double-buffered) --------
    if (wv == 0) tdm_load_2d_bf16(&Ktile[0][0], kbase, 32 * HD, 0, 32 * HD);
    __builtin_amdgcn_s_wait_tensorcnt(0);
    __syncthreads();
    for (int it = 0; it < NT; ++it) {
        int r0 = it * 32;
        if (it + 1 < NT && wv == 0)
            tdm_load_2d_bf16(&Ktile[(it + 1) & 1][0], kbase + (size_t)(r0 + 32) * HD,
                             32 * HD, 0, 32 * HD);
        const __bf16* kt = &Ktile[it & 1][0];
#pragma unroll
        for (int sub = 0; sub < 2; ++sub) {
            const __bf16* krow = kt + (sub * 16 + n15) * HD;
            bf16x16 b0 = frag_from_bf16(krow, half);
            bf16x16 b1 = frag_from_bf16(krow + 32, half);
            f32x8 sc = zero8();
            sc = wmma_bf16(aq0, b0, sc);
            sc = wmma_bf16(aq1, b1, sc);
            float mk = mkbase[r0 + sub * 16 + n15];
#pragma unroll
            for (int v = 0; v < 8; ++v) {
                float val = (mk != 0.0f) ? sc[v] * SCALE_INV : NEG_INF_SCORE;
                float tm = val;
#pragma unroll
                for (int d2 = 1; d2 < 16; d2 <<= 1) tm = fmaxf(tm, __shfl_xor(tm, d2, 32));
                float mn = fmaxf(m[v], tm);
                float e = __expf(val - mn);
#pragma unroll
                for (int d2 = 1; d2 < 16; d2 <<= 1) e += __shfl_xor(e, d2, 32);
                s[v] = s[v] * __expf(m[v] - mn) + e;
                m[v] = mn;
            }
        }
        __builtin_amdgcn_s_wait_tensorcnt(0);
        __syncthreads();
    }
    float rinv[8];
#pragma unroll
    for (int v = 0; v < 8; ++v) rinv[v] = 1.0f / s[v];

    // -------- pass 2: exact attn, importance, PV (K+V double-buffered) --------
    f32x8 ctxa[4];
#pragma unroll
    for (int t = 0; t < 4; ++t) ctxa[t] = zero8();
    __bf16* ps = &pstage[wv][0];

    if (wv == 0) {
        tdm_load_2d_bf16(&Ktile[0][0], kbase, 32 * HD, 0, 32 * HD);
        tdm_load_2d_bf16(&Vtile[0][0], vbase, 32, 64, R_PAD);
    }
    __builtin_amdgcn_s_wait_tensorcnt(0);
    __syncthreads();
    for (int it = 0; it < NT; ++it) {
        int r0 = it * 32;
        if (it + 1 < NT && wv == 0) {
            tdm_load_2d_bf16(&Ktile[(it + 1) & 1][0], kbase + (size_t)(r0 + 32) * HD,
                             32 * HD, 0, 32 * HD);
            tdm_load_2d_bf16(&Vtile[(it + 1) & 1][0], vbase + r0 + 32, 32, 64, R_PAD);
        }
        const __bf16* kt = &Ktile[it & 1][0];
        const __bf16* vt = &Vtile[it & 1][0];
#pragma unroll
        for (int sub = 0; sub < 2; ++sub) {
            const __bf16* krow = kt + (sub * 16 + n15) * HD;
            bf16x16 b0 = frag_from_bf16(krow, half);
            bf16x16 b1 = frag_from_bf16(krow + 32, half);
            f32x8 sc = zero8();
            sc = wmma_bf16(aq0, b0, sc);
            sc = wmma_bf16(aq1, b1, sc);
            float mk = mkbase[r0 + sub * 16 + n15];
            float colsum = 0.0f;
#pragma unroll
            for (int v = 0; v < 8; ++v) {
                float val = (mk != 0.0f) ? sc[v] * SCALE_INV : NEG_INF_SCORE;
                float p = __expf(val - m[v]) * rinv[v];
                ps[(v + 8 * half) * 32 + sub * 16 + n15] = (__bf16)p;
                colsum += p;
            }
            colsum += __shfl_xor(colsum, 16, 32);
            if (half == 0) atomicAdd(&imp[(size_t)h * R_PAD + r0 + sub * 16 + n15], colsum);
        }
        // PV: A = attn tile 16x32 (LDS restage), B = V^T rows (contiguous r)
        bf16x16 pa = frag_from_bf16(ps + n15 * 32, half);
#pragma unroll
        for (int dt = 0; dt < 4; ++dt) {
            bf16x16 bv = frag_from_bf16(vt + (dt * 16 + n15) * 32, half);
            ctxa[dt] = wmma_bf16(pa, bv, ctxa[dt]);
        }
        __builtin_amdgcn_s_wait_tensorcnt(0);
        __syncthreads();
    }
#pragma unroll
    for (int dt = 0; dt < 4; ++dt) {
#pragma unroll
        for (int v = 0; v < 8; ++v) {
            int row = q0 + v + 8 * half;
            ctx[(size_t)row * D_MODEL + h * HD + dt * 16 + n15] = (__bf16)ctxa[dt][v];
        }
    }
}

// Exact top-k by rank counting (value desc, index asc) — matches jax.lax.top_k.
__global__ __launch_bounds__(256) void k_topk(const float* __restrict__ imp,
                                              int* __restrict__ nt) {
    __shared__ float simp[R_IN];
    int h = blockIdx.x;
    const float* ih = imp + (size_t)h * R_PAD;
    for (int i = threadIdx.x; i < R_IN; i += blockDim.x) simp[i] = ih[i];
    __syncthreads();
    for (int i = threadIdx.x; i < R_IN; i += blockDim.x) {
        float vi = simp[i];
        int cnt = 0;
        for (int j = 0; j < R_IN; ++j) {
            float vj = simp[j];
            cnt += (vj > vi) || (vj == vi && j < i);
        }
        if (cnt < R_OUT_N) nt[(size_t)h * R_OUT_N + cnt] = i;
    }
}

// ---------------- launch ----------------
extern "C" void kernel_launch(void* const* d_in, const int* in_sizes, int n_in,
                              void* d_out, int out_size, void* d_ws, size_t ws_size,
                              hipStream_t stream) {
    (void)in_sizes; (void)n_in; (void)out_size; (void)ws_size;
    const float* x    = (const float*)d_in[0];
    const int*   mask = (const int*)d_in[1];
    const int*   topk = (const int*)d_in[2];
    const float* Wq = (const float*)d_in[3];
    const float* bq = (const float*)d_in[4];
    const float* Wk = (const float*)d_in[5];
    const float* bk = (const float*)d_in[6];
    const float* Wv = (const float*)d_in[7];
    const float* bv = (const float*)d_in[8];
    const float* Wo = (const float*)d_in[9];
    const float* bo = (const float*)d_in[10];

    char* ws = (char*)d_ws;
    size_t off = 0;
    auto carve = [&](size_t bytes) { char* p = ws + off; off = (off + bytes + 255) & ~(size_t)255; return p; };
    __bf16* Xb   = (__bf16*)carve((size_t)S_LEN * D_MODEL * 2);
    __bf16* Wqb  = (__bf16*)carve((size_t)D_MODEL * D_MODEL * 2);
    __bf16* Wkb  = (__bf16*)carve((size_t)D_MODEL * D_MODEL * 2);
    __bf16* Wvb  = (__bf16*)carve((size_t)D_MODEL * D_MODEL * 2);
    __bf16* Wob  = (__bf16*)carve((size_t)D_MODEL * D_MODEL * 2);
    __bf16* Qb   = (__bf16*)carve((size_t)S_LEN * D_MODEL * 2);
    __bf16* Kb   = (__bf16*)carve((size_t)S_LEN * D_MODEL * 2);
    __bf16* Vb   = (__bf16*)carve((size_t)S_LEN * D_MODEL * 2);
    __bf16* Ctx  = (__bf16*)carve((size_t)S_LEN * D_MODEL * 2);
    __bf16* Kp   = (__bf16*)carve((size_t)NHEAD * R_PAD * HD * 2);
    __bf16* VpT  = (__bf16*)carve((size_t)NHEAD * R_PAD * HD * 2);
    float*  mok  = (float*)carve((size_t)NHEAD * R_PAD * 4);
    float*  imp  = (float*)carve((size_t)NHEAD * R_PAD * 4);

    float* out_f = (float*)d_out;
    int*   out_i = (int*)(out_f + (size_t)S_LEN * D_MODEL);

    k_zero_f32<<<(NHEAD * R_PAD + 255) / 256, 256, 0, stream>>>(imp, NHEAD * R_PAD);

    int nx4 = S_LEN * D_MODEL / 4, nw4 = D_MODEL * D_MODEL / 4;
    k_cvt_bf16<<<(nx4 + 255) / 256, 256, 0, stream>>>(x, Xb, nx4);
    k_cvt_bf16<<<(nw4 + 255) / 256, 256, 0, stream>>>(Wq, Wqb, nw4);
    k_cvt_bf16<<<(nw4 + 255) / 256, 256, 0, stream>>>(Wk, Wkb, nw4);
    k_cvt_bf16<<<(nw4 + 255) / 256, 256, 0, stream>>>(Wv, Wvb, nw4);
    k_cvt_bf16<<<(nw4 + 255) / 256, 256, 0, stream>>>(Wo, Wob, nw4);

    dim3 gemm_grid(D_MODEL / 64, S_LEN / 256);
    k_gemm_qkv<<<gemm_grid, 256, 0, stream>>>(Xb, Wqb, bq, Qb);
    k_gemm_qkv<<<gemm_grid, 256, 0, stream>>>(Xb, Wkb, bk, Kb);
    k_gemm_qkv<<<gemm_grid, 256, 0, stream>>>(Xb, Wvb, bv, Vb);

    k_gather<<<(NHEAD * R_PAD + 255) / 256, 256, 0, stream>>>(Kb, Vb, topk, mask, Kp, VpT, mok);

    k_attn<<<NHEAD * 32, 256, 0, stream>>>(Qb, Kp, VpT, mok, imp, Ctx);

    k_gemm_out<<<gemm_grid, 256, 0, stream>>>(Ctx, Wob, bo, out_f);

    k_topk<<<NHEAD, 256, 0, stream>>>(imp, out_i);
}
